// LayerWiseSelect_13700945674792
// MI455X (gfx1250) — matrix operand order repaired and checked by reading, not compiled, not run on hardware
//
#include <hip/hip_runtime.h>
#include <hip/hip_bf16.h>
#include <math.h>

typedef __bf16 bf16_t;
typedef __attribute__((ext_vector_type(16))) __bf16 v16bf;
typedef __attribute__((ext_vector_type(8)))  __bf16 v8bf;
typedef __attribute__((ext_vector_type(4)))  __bf16 v4bf;
typedef __attribute__((ext_vector_type(8)))  float  v8f;
typedef __attribute__((ext_vector_type(4)))  float  v4f;

#define B_      8
#define L_      13
#define T_      1500
#define D_      1024
#define NSEL_   3
#define NUM_MASK 750
#define MTILES32 47         // ceil(1500/32)
#define STRIDE_A 1032       // bf16 elems per LDS row (pad 8 -> 4-bank row skew)

#define WT_BYTES (NSEL_ * D_ * D_ * 2)      // 6 MB bf16
#define S_BYTES  (147456)                   // 24*1500*4 rounded up

// ---------------------------------------------------------------------------
// Kernel 0: W[l][d][e] f32  ->  Wt[l][e][d] bf16   (one-time pack, 12 MB read)
// ---------------------------------------------------------------------------
__global__ __launch_bounds__(256) void lws_pack_w(const float* __restrict__ W,
                                                  bf16_t* __restrict__ Wt) {
    int idx = blockIdx.x * 256 + threadIdx.x;          // (l*D + e)*D + d
    if (idx >= NSEL_ * D_ * D_) return;
    int d = idx & (D_ - 1);
    int e = (idx >> 10) & (D_ - 1);
    int l = idx >> 20;
    Wt[idx] = (bf16_t)W[((size_t)l * D_ + d) * D_ + e];
}

// ---------------------------------------------------------------------------
// Kernel 1: s[b,l,t] = v[l] . tanh(x[b,l,t,:] @ W[l] + b[l]) + vb[l]
// Block: 256 thr = 8 waves, one 32-row tile of t staged in LDS as bf16.
// Wave w covers e in [w*128, w*128+128) as 8 N-tiles x 2 M-tiles; each B
// fragment load feeds two v_wmma_f32_16x16x32_bf16.
// ---------------------------------------------------------------------------
__global__ __launch_bounds__(256) void lws_score(const float* __restrict__ x,
                                                 const bf16_t* __restrict__ Wt,
                                                 const float* __restrict__ bias,
                                                 const float* __restrict__ vvec,
                                                 const float* __restrict__ vb,
                                                 float* __restrict__ s_out) {
    __shared__ __align__(16) bf16_t Asm[32 * STRIDE_A];   // ~66 KB
    __shared__ float s_lds[32];
    const int sel[NSEL_] = {5, 6, 7};

    const int mtile = blockIdx.x;          // 0..46 (32 rows each)
    const int lsel  = blockIdx.y;          // 0..2
    const int b     = blockIdx.z;          // 0..7
    const int layer = sel[lsel];

    const int tid  = threadIdx.x;
    const int wave = tid >> 5;
    const int lane = tid & 31;
    const int col  = lane & 15;            // column inside 16x16 tile / A row
    const int half = lane >> 4;            // 0 or 1

    const int t0 = mtile * 32;

    // ---- stage A tile (32 rows x 1024) as bf16 into LDS, row stride 1032 ----
    {
        const float* xbase = x + (((size_t)b * L_ + layer) * T_) * D_;
        for (int c = tid; c < 32 * (D_ / 4); c += 256) {
            int row = c >> 8;                      // 256 float4-chunks per row
            int cp  = (c & 255) * 4;
            int tr  = t0 + row;
            if (tr >= T_) tr = T_ - 1;             // clamp; padding rows unused
            v4f f = *(const v4f*)(xbase + (size_t)tr * D_ + cp);
            v4bf o;
#pragma unroll
            for (int j = 0; j < 4; ++j) o[j] = (bf16_t)f[j];
            *(v4bf*)(Asm + row * STRIDE_A + cp) = o;
        }
    }
    if (tid < 32) s_lds[tid] = 0.0f;
    __syncthreads();

    v8f acc0[8] = {};                      // M-tile 0 (rows t0..t0+15)
    v8f acc1[8] = {};                      // M-tile 1 (rows t0+16..t0+31)

    const int e_base = wave * 128;
    const int ecol   = e_base + col;       // e for n-tile 0; +n*16 per tile
    const bf16_t* wt_lane = Wt + ((size_t)lsel * D_ + ecol) * D_ + half * 16;

    // A fragment base pointers in LDS (this code's contiguous k order, same as B: lane row = col,
    // K = half*16 + j)
    const bf16_t* a0p = Asm + col * STRIDE_A + half * 16;
    const bf16_t* a1p = Asm + (col + 16) * STRIDE_A + half * 16;

    for (int k = 0; k < D_; k += 32) {
        v8bf lo0 = *(const v8bf*)(a0p + k);
        v8bf hi0 = *(const v8bf*)(a0p + k + 8);
        v8bf lo1 = *(const v8bf*)(a1p + k);
        v8bf hi1 = *(const v8bf*)(a1p + k + 8);
        v16bf af0 = __builtin_shufflevector(lo0, hi0,
                        0,1,2,3,4,5,6,7,8,9,10,11,12,13,14,15);
        v16bf af1 = __builtin_shufflevector(lo1, hi1,
                        0,1,2,3,4,5,6,7,8,9,10,11,12,13,14,15);
#pragma unroll
        for (int n = 0; n < 8; ++n) {
            // B fragment: col = ecol+n*16, K = half*16 + j -> 32B contiguous
            v16bf bfrag = *(const v16bf*)(wt_lane + (size_t)n * 16 * D_ + k);
            acc0[n] = __builtin_amdgcn_wmma_f32_16x16x32_bf16(
                false, af0, false, bfrag, (short)0, acc0[n], false, false);
            acc1[n] = __builtin_amdgcn_wmma_f32_16x16x32_bf16(
                false, af1, false, bfrag, (short)0, acc1[n], false, false);
        }
    }

    // ---- epilogue: tanh + dot with v, accumulated per row of the tiles ----
    float acc_s[2][8];
#pragma unroll
    for (int m = 0; m < 2; ++m)
#pragma unroll
        for (int r = 0; r < 8; ++r) acc_s[m][r] = 0.0f;

#pragma unroll
    for (int n = 0; n < 8; ++n) {
        int   e  = ecol + n * 16;
        float be = bias[lsel * D_ + e];
        float ve = vvec[lsel * D_ + e];
#pragma unroll
        for (int r = 0; r < 8; ++r) {      // D layout: row = half*8 + r, col = lane&15
            acc_s[0][r] += tanhf(acc0[n][r] + be) * ve;
            acc_s[1][r] += tanhf(acc1[n][r] + be) * ve;
        }
    }
    // reduce across the 16 columns (lanes 0-15 hold rows 0-7, 16-31 rows 8-15)
#pragma unroll
    for (int m = 0; m < 2; ++m)
#pragma unroll
        for (int r = 0; r < 8; ++r) {
            float vsum = acc_s[m][r];
            vsum += __shfl_xor(vsum, 8, 32);
            vsum += __shfl_xor(vsum, 4, 32);
            vsum += __shfl_xor(vsum, 2, 32);
            vsum += __shfl_xor(vsum, 1, 32);
            acc_s[m][r] = vsum;
        }
    if (col == 0) {                        // lane 0 -> rows 0-7, lane 16 -> rows 8-15
#pragma unroll
        for (int m = 0; m < 2; ++m)
#pragma unroll
            for (int r = 0; r < 8; ++r)
                atomicAdd(&s_lds[m * 16 + half * 8 + r], acc_s[m][r]);
    }
    __syncthreads();
    if (tid < 32 && (t0 + tid) < T_) {
        s_out[((size_t)b * NSEL_ + lsel) * T_ + t0 + tid] = s_lds[tid] + vb[lsel];
    }
}

// ---------------------------------------------------------------------------
// Kernel 2: top-750 mask (stable-argsort ranks via counting) + softmax -> attn
// ---------------------------------------------------------------------------
__global__ __launch_bounds__(256) void lws_attn(const float* __restrict__ s_in,
                                                float* __restrict__ attn) {
    __shared__ float sh[T_];
    __shared__ float red[256];
    const int bl  = blockIdx.x;            // 0..23 = b*3 + l
    const int tid = threadIdx.x;
    const float* s = s_in + (size_t)bl * T_;

    for (int i = tid; i < T_; i += 256) sh[i] = s[i];
    __syncthreads();

    float myv[6];
    int   mykeep[6];
    float lmax = -INFINITY;
#pragma unroll
    for (int c = 0; c < 6; ++c) {
        int tdx   = tid + c * 256;
        mykeep[c] = 0;
        myv[c]    = -INFINITY;
        if (tdx < T_) {
            float st  = sh[tdx];
            int   cnt = 0;
            for (int j = 0; j < T_; ++j) {
                float sj = sh[j];
                cnt += (sj < st) || (sj == st && j < tdx);   // stable ascending rank
            }
            if (cnt >= NUM_MASK) {
                mykeep[c] = 1;
                myv[c]    = st;
                lmax      = fmaxf(lmax, st);
            }
        }
    }
    red[tid] = lmax;
    __syncthreads();
    for (int off = 128; off > 0; off >>= 1) {
        if (tid < off) red[tid] = fmaxf(red[tid], red[tid + off]);
        __syncthreads();
    }
    float gmax = red[0];
    __syncthreads();

    float lsum = 0.0f;
#pragma unroll
    for (int c = 0; c < 6; ++c)
        if (mykeep[c]) lsum += expf(myv[c] - gmax);
    red[tid] = lsum;
    __syncthreads();
    for (int off = 128; off > 0; off >>= 1) {
        if (tid < off) red[tid] += red[tid + off];
        __syncthreads();
    }
    float inv = 1.0f / red[0];
#pragma unroll
    for (int c = 0; c < 6; ++c) {
        int tdx = tid + c * 256;
        if (tdx < T_)
            attn[(size_t)bl * T_ + tdx] = mykeep[c] ? expf(myv[c] - gmax) * inv : 0.0f;
    }
}

// ---------------------------------------------------------------------------
// Kernel 3: zero output (harness poisons d_out)
// ---------------------------------------------------------------------------
__global__ void lws_zero(float* __restrict__ out) {
    int i = blockIdx.x * 256 + threadIdx.x;
    if (i < B_ * D_) out[i] = 0.0f;
}

// ---------------------------------------------------------------------------
// Kernel 4: out[b,d] = (1/T) * sum_l lw[sel[l]] * sum_t attn[b,l,t]*x[b,l,t,d]
// Second streaming pass over x; masked (attn==0) rows are skipped uniformly.
// ---------------------------------------------------------------------------
__global__ __launch_bounds__(256) void lws_accum(const float* __restrict__ x,
                                                 const float* __restrict__ attn,
                                                 const float* __restrict__ lwts,
                                                 float* __restrict__ out) {
    const int sel[NSEL_] = {5, 6, 7};
    const int b      = blockIdx.z;         // 0..7
    const int dchunk = blockIdx.y;         // 0..3
    const int tchunk = blockIdx.x;         // 0..9 (150 t each)
    const int d      = dchunk * 256 + threadIdx.x;

    float acc = 0.0f;
    const int tbeg = tchunk * 150;
    const int tend = tbeg + 150;
#pragma unroll
    for (int l = 0; l < NSEL_; ++l) {
        float lw = lwts[sel[l]];
        const float* xb = x + (((size_t)b * L_ + sel[l]) * T_) * D_;
        const float* ab = attn + ((size_t)b * NSEL_ + l) * T_;
        for (int tt = tbeg; tt < tend; ++tt) {
            float w = ab[tt] * lw;
            if (w != 0.0f) acc += w * xb[(size_t)tt * D_ + d];
        }
    }
    atomicAdd(&out[b * D_ + d], acc * (1.0f / (float)T_));
}

// ---------------------------------------------------------------------------
extern "C" void kernel_launch(void* const* d_in, const int* in_sizes, int n_in,
                              void* d_out, int out_size, void* d_ws, size_t ws_size,
                              hipStream_t stream) {
    const float* layer_reps = (const float*)d_in[0];   // [8,13,1500,1024]
    const float* W          = (const float*)d_in[1];   // [3,1024,1024]
    const float* bvec       = (const float*)d_in[2];   // [3,1024]
    const float* vvec       = (const float*)d_in[3];   // [3,1024]
    const float* vb         = (const float*)d_in[4];   // [3]
    const float* lwts       = (const float*)d_in[5];   // [13]
    float* out = (float*)d_out;                        // [8,1024]

    char*   ws    = (char*)d_ws;
    bf16_t* Wt    = (bf16_t*)ws;                               // 6 MB
    float*  s_buf = (float*)(ws + WT_BYTES);                   // 24*1500 f32
    float*  attn  = (float*)(ws + WT_BYTES + S_BYTES);         // 24*1500 f32

    lws_pack_w<<<(NSEL_ * D_ * D_) / 256, 256, 0, stream>>>(W, Wt);
    lws_score<<<dim3(MTILES32, NSEL_, B_), 256, 0, stream>>>(layer_reps, Wt, bvec,
                                                             vvec, vb, s_buf);
    lws_attn<<<B_ * NSEL_, 256, 0, stream>>>(s_buf, attn);
    lws_zero<<<(B_ * D_ + 255) / 256, 256, 0, stream>>>(out);
    lws_accum<<<dim3(10, 4, B_), 256, 0, stream>>>(layer_reps, attn, lwts, out);
}
